// MultiHeadSelfAtt_49546742726697
// MI455X (gfx1250) — matrix operand order repaired and checked by reading, not compiled, not run on hardware
//
#include <hip/hip_runtime.h>
#include <hip/hip_bf16.h>

typedef __attribute__((ext_vector_type(16))) _Float16 v16h;
typedef __attribute__((ext_vector_type(8)))  _Float16 v8h;
typedef __attribute__((ext_vector_type(4)))  _Float16 v4h;
typedef __attribute__((ext_vector_type(8)))  float    v8f;

union F16x16 { v16h v; v8h h[2]; };

__device__ __forceinline__ v8f wmma_f16(const F16x16& a, const F16x16& b, v8f c) {
    // D = A(16x32 f16) * B(32x16 f16) + C(16x16 f32)
    return __builtin_amdgcn_wmma_f32_16x16x32_f16(
        /*neg_a=*/false, a.v, /*neg_b=*/false, b.v,
        /*c_mod=*/(short)0, c, /*reuse_a=*/false, /*reuse_b=*/false);
}

// async copy 4 bytes per lane: global -> LDS, tracked by ASYNCcnt
__device__ __forceinline__ void stage_async_b32(const _Float16* gsrc, _Float16* lds_dst) {
    // low 32 bits of a flat shared-aperture address are the LDS byte offset
    unsigned off = (unsigned)(unsigned long long)lds_dst;
    asm volatile("global_load_async_to_lds_b32 %0, %1, off"
                 :: "v"(off), "v"(gsrc) : "memory");
}
__device__ __forceinline__ void wait_async_le1() {
    asm volatile("s_wait_asynccnt 0x1" ::: "memory");
}
__device__ __forceinline__ void wait_async_0() {
    asm volatile("s_wait_asynccnt 0x0" ::: "memory");
}

// ---------------------------------------------------------------------------
// f32 -> f16 conversion
// ---------------------------------------------------------------------------
__global__ void __launch_bounds__(256)
cvt_kernel(const float* __restrict__ src, _Float16* __restrict__ dst, int n) {
    int i = blockIdx.x * 256 + threadIdx.x;
    if (i < n) dst[i] = (_Float16)src[i];
}

// ---------------------------------------------------------------------------
// Cooperative GEMM body: block(256) computes 16(M) x 512(N).
// A k-chunk (16x32 f16 = 1KB) double-buffered in LDS via async-to-LDS copies;
// each of 8 waves owns a 16x64 slice of N, B-frags straight from global.
// ---------------------------------------------------------------------------
__device__ __forceinline__ void gemm_tile_16x512(
        const _Float16* __restrict__ X, const _Float16* __restrict__ W,
        int m0, int n0, int lane, _Float16* Atile0, _Float16* Atile1,
        v8f acc[4]) {
    const int tid = threadIdx.x;
    const int c16 = lane & 15, g = lane >> 4;

    // staging assignment: thread tid moves 4 bytes of the 16x32 A chunk
    const int arow = tid >> 4;            // 0..15
    const int acolh = (tid & 15) * 2;     // half index within the 32-wide chunk
    const _Float16* agsrc = X + (size_t)(m0 + arow) * 1024 + acolh;
    _Float16* adst0 = Atile0 + tid * 2;
    _Float16* adst1 = Atile1 + tid * 2;

    stage_async_b32(agsrc, adst0);                        // prologue: chunk 0

    for (int i = 0; i < 32; ++i) {
        const int k0 = i * 32;
        if (i + 1 < 32) {
            stage_async_b32(agsrc + (i + 1) * 32, ((i + 1) & 1) ? adst1 : adst0);
            wait_async_le1();                             // chunk i landed
        } else {
            wait_async_0();
        }
        __syncthreads();                                  // tile visible to all waves

        const _Float16* ab = (i & 1) ? Atile1 : Atile0;
        F16x16 a;                                         // A frag from LDS
        a.h[0] = *(const v8h*)(ab + c16 * 32 + g * 8);
        a.h[1] = *(const v8h*)(ab + c16 * 32 + 16 + g * 8);
#pragma unroll
        for (int nt = 0; nt < 4; ++nt) {                  // B: 32x16, B[k][n]=W[n][k]
            const _Float16* wrow =
                W + (size_t)(n0 + nt * 16 + c16) * 1024 + k0 + g * 8;
            __builtin_prefetch(wrow + 32, 0, 1);          // next k-chunk of this row
            F16x16 bfr;
            bfr.h[0] = *(const v8h*)(wrow);
            bfr.h[1] = *(const v8h*)(wrow + 16);
            acc[nt] = wmma_f16(a, bfr, acc[nt]);
        }
        __syncthreads();                                  // reads done before re-stage
    }
}

// QKV projection: Out = X(4096x1024,f16) * W^T + b ; blockIdx.y selects Q/K/V
__global__ void __launch_bounds__(256)
gemm_qkv_kernel(const _Float16* __restrict__ X,
                const _Float16* __restrict__ W0, const _Float16* __restrict__ W1,
                const _Float16* __restrict__ W2,
                const float* __restrict__ b0, const float* __restrict__ b1,
                const float* __restrict__ b2,
                _Float16* __restrict__ O0, _Float16* __restrict__ O1,
                _Float16* __restrict__ O2) {
    __shared__ __align__(16) _Float16 Atile[2][16 * 32];
    const int lane = threadIdx.x & 31;
    const int wave = threadIdx.x >> 5;
    const int which = blockIdx.y;
    const _Float16* W  = (which == 0) ? W0 : (which == 1) ? W1 : W2;
    const float* bias  = (which == 0) ? b0 : (which == 1) ? b1 : b2;
    _Float16* Out      = (which == 0) ? O0 : (which == 1) ? O1 : O2;

    const int m0 = (blockIdx.x >> 1) * 16;                // 256 m-tiles
    const int n0 = (blockIdx.x & 1) * 512 + wave * 64;    // wave's N slice
    const int c16 = lane & 15, g = lane >> 4;

    v8f acc[4] = {};
    gemm_tile_16x512(X, W, m0, n0, lane, Atile[0], Atile[1], acc);

#pragma unroll
    for (int nt = 0; nt < 4; ++nt) {
        const int col = n0 + nt * 16 + c16;
        const float bj = bias[col];
#pragma unroll
        for (int r = 0; r < 8; ++r) {                     // C layout: row = r + g*8
            const int orow = m0 + r + g * 8;
            Out[(size_t)orow * 1024 + col] = (_Float16)(acc[nt][r] + bj);
        }
    }
}

// WL projection: Y = X2h * WL^T + bL  (f32 out, relu applied later)
__global__ void __launch_bounds__(256)
gemm_wl_kernel(const _Float16* __restrict__ X, const _Float16* __restrict__ W,
               const float* __restrict__ bias, float* __restrict__ Out) {
    __shared__ __align__(16) _Float16 Atile[2][16 * 32];
    const int lane = threadIdx.x & 31;
    const int wave = threadIdx.x >> 5;
    const int m0 = (blockIdx.x >> 1) * 16;
    const int n0 = (blockIdx.x & 1) * 512 + wave * 64;
    const int c16 = lane & 15, g = lane >> 4;

    v8f acc[4] = {};
    gemm_tile_16x512(X, W, m0, n0, lane, Atile[0], Atile[1], acc);

#pragma unroll
    for (int nt = 0; nt < 4; ++nt) {
        const int col = n0 + nt * 16 + c16;
        const float bj = bias[col];
#pragma unroll
        for (int r = 0; r < 8; ++r) {
            const int orow = m0 + r + g * 8;
            Out[(size_t)orow * 1024 + col] = acc[nt][r] + bj;
        }
    }
}

// ---------------------------------------------------------------------------
// Attention: att[l, h*64 + i] = sum_m softmax_m( K[l]·Q[m] / 8 ) * V[m, i]
// (reference's "softmax over query axis" == flash attention with Q/K swapped)
// one wave per (batch, head, 16-row l-tile); two-pass (row-max, then exp+PV)
// ---------------------------------------------------------------------------
__global__ void __launch_bounds__(32)
attn_kernel(const _Float16* __restrict__ Q, const _Float16* __restrict__ K,
            const _Float16* __restrict__ V, float* __restrict__ att) {
    __shared__ __align__(16) _Float16 Pbuf[16 * 32];   // P tile, row-major [l][m']
    __shared__ __align__(16) _Float16 Vt[64 * 32];     // V^T tile  [i][m']

    const int lane = threadIdx.x & 31;
    const int c16 = lane & 15, g = lane >> 4;
    const int l0 = blockIdx.x * 16;          // l-tile within batch
    const int h  = blockIdx.y;
    const size_t seq = (size_t)blockIdx.z * 2048;

    // A-frags: 16 K-rows x 64 i, as two 16x32 chunks
    F16x16 aK[2];
    {
        const _Float16* krow = K + (seq + l0 + c16) * 1024 + h * 64;
#pragma unroll
        for (int c = 0; c < 2; ++c) {
            aK[c].h[0] = *(const v8h*)(krow + c * 32 + g * 8);
            aK[c].h[1] = *(const v8h*)(krow + c * 32 + 16 + g * 8);
        }
    }

    // ---- pass 1: row max of raw scores -----------------------------------
    float rmax[8];
#pragma unroll
    for (int r = 0; r < 8; ++r) rmax[r] = -1e30f;
    for (int mt = 0; mt < 128; ++mt) {
        const _Float16* qrow = Q + (seq + mt * 16 + c16) * 1024 + h * 64;
        v8f s = {};
#pragma unroll
        for (int c = 0; c < 2; ++c) {        // B[k=i][n=m] = Q[m][i]
            F16x16 bq;
            bq.h[0] = *(const v8h*)(qrow + c * 32 + g * 8);
            bq.h[1] = *(const v8h*)(qrow + c * 32 + 16 + g * 8);
            s = wmma_f16(aK[c], bq, s);
        }
#pragma unroll
        for (int r = 0; r < 8; ++r) rmax[r] = fmaxf(rmax[r], s[r]);
    }
#pragma unroll
    for (int r = 0; r < 8; ++r)
#pragma unroll
        for (int m = 8; m >= 1; m >>= 1)     // reduce over 16-lane half (cols)
            rmax[r] = fmaxf(rmax[r], __shfl_xor(rmax[r], m, 32));

    // ---- pass 2: exp, row-sum, O += P * V --------------------------------
    v8f o[4] = {};
    float rsum[8] = {};
    for (int ch = 0; ch < 64; ++ch) {
        const int m0c = ch * 32;
#pragma unroll
        for (int sub = 0; sub < 2; ++sub) {
            const _Float16* qrow = Q + (seq + m0c + sub * 16 + c16) * 1024 + h * 64;
            v8f s = {};
#pragma unroll
            for (int c = 0; c < 2; ++c) {
                F16x16 bq;
                bq.h[0] = *(const v8h*)(qrow + c * 32 + g * 8);
                bq.h[1] = *(const v8h*)(qrow + c * 32 + 16 + g * 8);
                s = wmma_f16(aK[c], bq, s);
            }
#pragma unroll
            for (int r = 0; r < 8; ++r) {
                const float p = __expf((s[r] - rmax[r]) * 0.125f);
                rsum[r] += p;
                Pbuf[(r + g * 8) * 32 + sub * 16 + c16] = (_Float16)p;
            }
        }
        // stage V chunk transposed: lane -> one m' row, write Vt[i][m']
        {
            const _Float16* vrow = V + (seq + m0c + lane) * 1024 + h * 64;
#pragma unroll
            for (int i0 = 0; i0 < 64; i0 += 8) {
                v8h vv = *(const v8h*)(vrow + i0);
#pragma unroll
                for (int j = 0; j < 8; ++j) Vt[(i0 + j) * 32 + lane] = vv[j];
            }
        }
        asm volatile("s_wait_dscnt 0" ::: "memory");  // LDS stores visible to frag reads
        F16x16 pf;                                     // A = P (16 x 32)
        pf.h[0] = *(const v8h*)(&Pbuf[c16 * 32 + g * 8]);
        pf.h[1] = *(const v8h*)(&Pbuf[c16 * 32 + 16 + g * 8]);
#pragma unroll
        for (int nt = 0; nt < 4; ++nt) {               // B = V chunk (32 x 16)
            const _Float16* vt = &Vt[(nt * 16 + c16) * 32 + g * 8];
            F16x16 bv;
            bv.h[0] = *(const v8h*)(vt);
            bv.h[1] = *(const v8h*)(vt + 16);
            o[nt] = wmma_f16(pf, bv, o[nt]);
        }
    }
#pragma unroll
    for (int r = 0; r < 8; ++r)
#pragma unroll
        for (int m = 8; m >= 1; m >>= 1)
            rsum[r] += __shfl_xor(rsum[r], m, 32);

    float* ob = att + (seq + l0) * 1024 + h * 64;
#pragma unroll
    for (int r = 0; r < 8; ++r) {
        const float inv = 1.0f / rsum[r];
        const int orow = r + g * 8;
#pragma unroll
        for (int nt = 0; nt < 4; ++nt)
            ob[(size_t)orow * 1024 + nt * 16 + c16] = o[nt][r] * inv;
    }
}

// ---------------------------------------------------------------------------
// row reductions for LayerNorm (256 threads, 8 waves)
// ---------------------------------------------------------------------------
__device__ __forceinline__ void block_reduce2(float& s, float& q, float* red) {
#pragma unroll
    for (int m = 16; m >= 1; m >>= 1) {
        s += __shfl_xor(s, m, 32);
        q += __shfl_xor(q, m, 32);
    }
    const int lane = threadIdx.x & 31, w = threadIdx.x >> 5;
    if (lane == 0) { red[w] = s; red[8 + w] = q; }
    __syncthreads();
    if (threadIdx.x == 0) {
        float ts = 0.f, tq = 0.f;
        for (int i = 0; i < 8; ++i) { ts += red[i]; tq += red[8 + i]; }
        red[16] = ts; red[17] = tq;
    }
    __syncthreads();
    s = red[16]; q = red[17];
    __syncthreads();
}

// x1 = x + LN(att); x2 = LN(x1); emits x2 (f32) and x2 (f16)
__global__ void __launch_bounds__(256)
ln_mid_kernel(const float* __restrict__ x, const float* __restrict__ att,
              const float* __restrict__ gamma, const float* __restrict__ beta,
              float* __restrict__ x2, _Float16* __restrict__ x2h) {
    __shared__ float red[18];
    const int t = threadIdx.x;
    const size_t base = (size_t)blockIdx.x * 1024;
    float4 a  = ((const float4*)(att + base))[t];
    float4 xr = ((const float4*)(x + base))[t];
    float4 gv = ((const float4*)gamma)[t];
    float4 bv = ((const float4*)beta)[t];

    float s = a.x + a.y + a.z + a.w;
    float q = a.x * a.x + a.y * a.y + a.z * a.z + a.w * a.w;
    block_reduce2(s, q, red);
    float mu = s * (1.f / 1024.f);
    float rs = rsqrtf(fmaxf(q * (1.f / 1024.f) - mu * mu, 0.f) + 1e-5f);

    float4 x1;
    x1.x = xr.x + (a.x - mu) * rs * gv.x + bv.x;
    x1.y = xr.y + (a.y - mu) * rs * gv.y + bv.y;
    x1.z = xr.z + (a.z - mu) * rs * gv.z + bv.z;
    x1.w = xr.w + (a.w - mu) * rs * gv.w + bv.w;

    s = x1.x + x1.y + x1.z + x1.w;
    q = x1.x * x1.x + x1.y * x1.y + x1.z * x1.z + x1.w * x1.w;
    block_reduce2(s, q, red);
    mu = s * (1.f / 1024.f);
    rs = rsqrtf(fmaxf(q * (1.f / 1024.f) - mu * mu, 0.f) + 1e-5f);

    float4 o;
    o.x = (x1.x - mu) * rs * gv.x + bv.x;
    o.y = (x1.y - mu) * rs * gv.y + bv.y;
    o.z = (x1.z - mu) * rs * gv.z + bv.z;
    o.w = (x1.w - mu) * rs * gv.w + bv.w;
    ((float4*)(x2 + base))[t] = o;
    v4h oh = {(_Float16)o.x, (_Float16)o.y, (_Float16)o.z, (_Float16)o.w};
    ((v4h*)(x2h + base))[t] = oh;
}

// out = LN(x2 + relu(y))
__global__ void __launch_bounds__(256)
ln_out_kernel(const float* __restrict__ x2, const float* __restrict__ y,
              const float* __restrict__ gamma, const float* __restrict__ beta,
              float* __restrict__ out) {
    __shared__ float red[18];
    const int t = threadIdx.x;
    const size_t base = (size_t)blockIdx.x * 1024;
    float4 xv = ((const float4*)(x2 + base))[t];
    float4 yv = ((const float4*)(y + base))[t];
    float4 gv = ((const float4*)gamma)[t];
    float4 bv = ((const float4*)beta)[t];

    float4 u;
    u.x = xv.x + fmaxf(yv.x, 0.f);
    u.y = xv.y + fmaxf(yv.y, 0.f);
    u.z = xv.z + fmaxf(yv.z, 0.f);
    u.w = xv.w + fmaxf(yv.w, 0.f);

    float s = u.x + u.y + u.z + u.w;
    float q = u.x * u.x + u.y * u.y + u.z * u.z + u.w * u.w;
    block_reduce2(s, q, red);
    float mu = s * (1.f / 1024.f);
    float rs = rsqrtf(fmaxf(q * (1.f / 1024.f) - mu * mu, 0.f) + 1e-5f);

    float4 o;
    o.x = (u.x - mu) * rs * gv.x + bv.x;
    o.y = (u.y - mu) * rs * gv.y + bv.y;
    o.z = (u.z - mu) * rs * gv.z + bv.z;
    o.w = (u.w - mu) * rs * gv.w + bv.w;
    ((float4*)(out + base))[t] = o;
}

// ---------------------------------------------------------------------------
extern "C" void kernel_launch(void* const* d_in, const int* in_sizes, int n_in,
                              void* d_out, int out_size, void* d_ws, size_t ws_size,
                              hipStream_t stream) {
    (void)in_sizes; (void)n_in; (void)out_size; (void)ws_size;
    const float* x  = (const float*)d_in[0];
    const float* WQ = (const float*)d_in[1];
    const float* bQ = (const float*)d_in[2];
    const float* WK = (const float*)d_in[3];
    const float* bK = (const float*)d_in[4];
    const float* WV = (const float*)d_in[5];
    const float* bV = (const float*)d_in[6];
    const float* WL = (const float*)d_in[7];
    const float* bL = (const float*)d_in[8];
    const float* g  = (const float*)d_in[9];
    const float* bb = (const float*)d_in[10];
    float* out = (float*)d_out;

    char* ws = (char*)d_ws;
    const size_t MB = 1u << 20;
    _Float16* XH  = (_Float16*)(ws + 0);        //  8 MB : x in f16
    _Float16* WQH = (_Float16*)(ws + 8 * MB);   //  2 MB
    _Float16* WKH = (_Float16*)(ws + 10 * MB);  //  2 MB
    _Float16* WVH = (_Float16*)(ws + 12 * MB);  //  2 MB
    _Float16* WLH = (_Float16*)(ws + 14 * MB);  //  2 MB
    _Float16* QH  = (_Float16*)(ws + 16 * MB);  //  8 MB
    _Float16* KH  = (_Float16*)(ws + 24 * MB);  //  8 MB
    _Float16* VH  = (_Float16*)(ws + 32 * MB);  //  8 MB
    float*    ATT = (float*)(ws + 40 * MB);     // 16 MB
    float*    X2  = (float*)(ws + 56 * MB);     // 16 MB
    _Float16* X2H = (_Float16*)(ws + 72 * MB);  //  8 MB
    float*    Y   = (float*)(ws + 16 * MB);     // 16 MB, aliases QH/KH (dead by then)

    // 1) converts to f16
    cvt_kernel<<<16384, 256, 0, stream>>>(x, XH, 4096 * 1024);
    cvt_kernel<<<4096, 256, 0, stream>>>(WQ, WQH, 1024 * 1024);
    cvt_kernel<<<4096, 256, 0, stream>>>(WK, WKH, 1024 * 1024);
    cvt_kernel<<<4096, 256, 0, stream>>>(WV, WVH, 1024 * 1024);
    cvt_kernel<<<4096, 256, 0, stream>>>(WL, WLH, 1024 * 1024);

    // 2) QKV projections (WMMA + async-to-LDS double buffering)
    gemm_qkv_kernel<<<dim3(512, 3), 256, 0, stream>>>(XH, WQH, WKH, WVH,
                                                      bQ, bK, bV, QH, KH, VH);
    // 3) attention (WMMA + LDS transpose staging)
    attn_kernel<<<dim3(128, 16, 2), 32, 0, stream>>>(QH, KH, VH, ATT);

    // 4) x1 = x + LN(att); x2 = LN(x1)
    ln_mid_kernel<<<4096, 256, 0, stream>>>(x, ATT, g, bb, X2, X2H);

    // 5) y = x2 @ WL^T + bL (WMMA + async-to-LDS double buffering)
    gemm_wl_kernel<<<512, 256, 0, stream>>>(X2H, WLH, bL, Y);

    // 6) out = LN(x2 + relu(y))
    ln_out_kernel<<<4096, 256, 0, stream>>>(X2, Y, g, bb, out);
}
